// GBottleneck_58291296141338
// MI455X (gfx1250) — matrix operand order repaired and checked by reading, not compiled, not run on hardware
//
#include <hip/hip_runtime.h>

typedef __attribute__((ext_vector_type(16))) __bf16 bf16x16;
typedef __attribute__((ext_vector_type(8)))  __bf16 bf16x8;
typedef __attribute__((ext_vector_type(4)))  __bf16 bf16x4;
typedef __attribute__((ext_vector_type(8)))  float  f32x8;
typedef __attribute__((ext_vector_type(4)))  float  f32x4;

#define NODES 8192
#define INDIM 256
#define HID   192
#define LBLK  6
#define KSPLIT 4

// ---------------- fragment loaders (bf16, wave32) ----------------
// A (16x32, MxK): lane L holds row M=L&15; half=L>>4:
//   elems 0..7  = K[half*8 .. +7],  elems 8..15 = K[16+half*8 .. +7]
__device__ __forceinline__ bf16x16 load_a(const __bf16* __restrict__ A, size_t lda,
                                          int m0, int k0, int lane) {
  const __bf16* p = A + (size_t)(m0 + (lane & 15)) * lda + (size_t)k0 + (size_t)((lane >> 4) * 8);
  bf16x8 lo = *(const bf16x8*)p;
  bf16x8 hi = *(const bf16x8*)(p + 16);
  return __builtin_shufflevector(lo, hi, 0,1,2,3,4,5,6,7,8,9,10,11,12,13,14,15);
}
// B (32x16, KxN) from BT stored N x K row-major: lane holds column N=L&15,
//   elems 0..7 = K[half*8 .. +7], elems 8..15 = K[16+half*8 .. +7]
__device__ __forceinline__ bf16x16 load_b(const __bf16* __restrict__ BT, size_t ldb,
                                          int n0, int k0, int lane) {
  const __bf16* p = BT + (size_t)(n0 + (lane & 15)) * ldb + (size_t)k0 + (size_t)((lane >> 4) * 8);
  bf16x8 lo = *(const bf16x8*)p;
  bf16x8 hi = *(const bf16x8*)(p + 16);
  return __builtin_shufflevector(lo, hi, 0,1,2,3,4,5,6,7,8,9,10,11,12,13,14,15);
}

__device__ __forceinline__ f32x8 wmma_bf16(bf16x16 a, bf16x16 b, f32x8 c) {
  return __builtin_amdgcn_wmma_f32_16x16x32_bf16(false, a, false, b, (short)0, c, false, false);
}

// ---------------- small GEMM (activations x weights) ----------------
// C[M x NT*16] = A[M x K] * BT[NT*16 x K]^T ; epilogue: (+bias)->outF fp32, or outBT bf16 transposed
template<int NT>
__launch_bounds__(256)
__global__ void gemm1_bf16(const __bf16* __restrict__ A,  int lda,
                           const __bf16* __restrict__ BT, int ldb,
                           const float*  __restrict__ bias,
                           float*   __restrict__ outF, int ldc,
                           __bf16*  __restrict__ outBT, int ldt,
                           int K)
{
  const int lane = threadIdx.x & 31;
  const int wave = threadIdx.x >> 5;
  const int m0 = blockIdx.x * 128 + wave * 16;
  const int n0 = blockIdx.y * (NT * 16);
  const int nl = lane & 15;
  const int mh = lane >> 4;

  f32x8 acc[NT];
  #pragma unroll
  for (int nt = 0; nt < NT; ++nt)
    #pragma unroll
    for (int r = 0; r < 8; ++r) acc[nt][r] = 0.0f;

  for (int k0 = 0; k0 < K; k0 += 32) {
    bf16x16 a = load_a(A, (size_t)lda, m0, k0, lane);
    #pragma unroll
    for (int nt = 0; nt < NT; ++nt) {
      bf16x16 b = load_b(BT, (size_t)ldb, n0 + nt * 16, k0, lane);
      acc[nt] = wmma_bf16(a, b, acc[nt]);
    }
  }

  #pragma unroll
  for (int nt = 0; nt < NT; ++nt) {
    #pragma unroll
    for (int r = 0; r < 8; ++r) {
      const int m  = m0 + mh * 8 + r;
      const int nn = n0 + nt * 16 + nl;
      float s = acc[nt][r];
      if (bias) s += bias[nn];
      if (outF)  outF [(size_t)m * ldc + nn] = s;
      if (outBT) outBT[(size_t)nn * ldt + m] = (__bf16)s;
    }
  }
}

// ---------------- big GEMM: split-K partial accumulation ----------------
// partial[z][M x ldp] += A[M, kz..kz+K/KS) * B ; wave tile = 32 rows x NT*16 cols
template<int NT>
__launch_bounds__(256)
__global__ void gemm2_part(const __bf16* __restrict__ A,  int lda,
                           const __bf16* __restrict__ BT, int ldb,
                           float* __restrict__ part, size_t partStride,
                           int ldp, int K)
{
  const int lane = threadIdx.x & 31;
  const int wave = threadIdx.x >> 5;
  const int m0 = blockIdx.x * 256 + wave * 32;       // 8 waves x 32 rows
  const int n0 = blockIdx.y * (NT * 16);
  const int z  = blockIdx.z;
  const int kb = (K / KSPLIT) * z;
  const int ke = kb + K / KSPLIT;
  const int nl = lane & 15;
  const int mh = lane >> 4;

  f32x8 acc[2][NT];
  #pragma unroll
  for (int mt = 0; mt < 2; ++mt)
    #pragma unroll
    for (int nt = 0; nt < NT; ++nt)
      #pragma unroll
      for (int r = 0; r < 8; ++r) acc[mt][nt][r] = 0.0f;

  #pragma unroll 2
  for (int k0 = kb; k0 < ke; k0 += 32) {
    bf16x16 a0 = load_a(A, (size_t)lda, m0,      k0, lane);
    bf16x16 a1 = load_a(A, (size_t)lda, m0 + 16, k0, lane);
    #pragma unroll
    for (int nt = 0; nt < NT; ++nt) {
      bf16x16 b = load_b(BT, (size_t)ldb, n0 + nt * 16, k0, lane);
      acc[0][nt] = wmma_bf16(a0, b, acc[0][nt]);
      acc[1][nt] = wmma_bf16(a1, b, acc[1][nt]);
    }
  }

  float* po = part + (size_t)z * partStride;
  #pragma unroll
  for (int mt = 0; mt < 2; ++mt)
    #pragma unroll
    for (int nt = 0; nt < NT; ++nt)
      #pragma unroll
      for (int r = 0; r < 8; ++r) {
        const int m  = m0 + mt * 16 + mh * 8 + r;
        const int nn = n0 + nt * 16 + nl;
        po[(size_t)m * ldp + nn] = acc[mt][nt][r];
      }
}

// ---------------- combine: sum partials + fused gconv epilogue ----------------
__global__ void k_combine(const float* __restrict__ part, size_t partStride,
                          const float* __restrict__ vC,
                          const float* __restrict__ hres,
                          float* __restrict__ outF,
                          __bf16* __restrict__ outB,
                          int total, int relu)
{
  int i = blockIdx.x * 256 + threadIdx.x;
  if (i >= total) return;
  float s = vC[i];
  #pragma unroll
  for (int z = 0; z < KSPLIT; ++z) s += part[(size_t)z * partStride + i];
  if (relu) s = fmaxf(s, 0.0f);
  if (hres) s = (hres[i] + s) * 0.5f;
  if (outF) outF[i] = s;
  if (outB) outB[i] = (__bf16)s;
}

// ---------------- helpers ----------------
__global__ void k_f32_to_bf16_x4(const float* __restrict__ src, __bf16* __restrict__ dst, long long n4) {
  long long i = (long long)blockIdx.x * 256 + threadIdx.x;
  if (i >= n4) return;
  f32x4 v = *(const f32x4*)(src + 4 * i);
  bf16x4 o;
  o[0] = (__bf16)v[0]; o[1] = (__bf16)v[1]; o[2] = (__bf16)v[2]; o[3] = (__bf16)v[3];
  *(bf16x4*)(dst + 4 * i) = o;
}

// dst (Npad x K, bf16) = transpose(src (K x Nsrc, f32)), zero-padded rows
__global__ void k_transposeW(const float* __restrict__ src, __bf16* __restrict__ dst,
                             int K, int Nsrc, int Npad) {
  int idx = blockIdx.x * 256 + threadIdx.x;
  if (idx >= K * Npad) return;
  int nn = idx / K, k = idx - nn * K;
  float v = (nn < Nsrc) ? src[(size_t)k * Nsrc + nn] : 0.0f;
  dst[(size_t)nn * K + k] = (__bf16)v;
}

__global__ void k_padbias(const float* __restrict__ src, float* __restrict__ dst, int nsrc, int npad) {
  int i = threadIdx.x;
  if (i < npad) dst[i] = (i < nsrc) ? src[i] : 0.0f;
}

__global__ void k_pack_out(const float* __restrict__ xpad, const float* __restrict__ h,
                           float* __restrict__ out) {
  int i = blockIdx.x * 256 + threadIdx.x;
  const int n1 = NODES * 3;
  const int n2 = NODES * HID;
  if (i < n1) {
    int r = i / 3, c = i - r * 3;
    out[i] = xpad[(size_t)r * 16 + c];
  } else if (i < n1 + n2) {
    out[i] = h[i - n1];
  }
}

// ---------------- launcher ----------------
extern "C" void kernel_launch(void* const* d_in, const int* in_sizes, int n_in,
                              void* d_out, int out_size, void* d_ws, size_t ws_size,
                              hipStream_t stream) {
  const float* x    = (const float*)d_in[0];
  const float* adj  = (const float*)d_in[1];
  const float* in_W = (const float*)d_in[2];
  const float* in_Wl= (const float*)d_in[3];
  const float* in_b = (const float*)d_in[4];
  const float* bW1  = (const float*)d_in[5];
  const float* bWl1 = (const float*)d_in[6];
  const float* bb1  = (const float*)d_in[7];
  const float* bW2  = (const float*)d_in[8];
  const float* bWl2 = (const float*)d_in[9];
  const float* bb2  = (const float*)d_in[10];
  const float* oW   = (const float*)d_in[11];
  const float* oWl  = (const float*)d_in[12];
  const float* ob   = (const float*)d_in[13];
  float* out = (float*)d_out;

  char* ws = (char*)d_ws;
  size_t off = 0;
  auto alloc = [&](size_t bytes) -> void* {
    void* p = ws + off;
    off += bytes;
    off = (off + 255) & ~(size_t)255;
    return p;
  };
  __bf16* adjb = (__bf16*)alloc((size_t)NODES * NODES * 2);   // 128 MB, L2-resident
  __bf16* xb   = (__bf16*)alloc((size_t)NODES * INDIM * 2);
  __bf16* hb   = (__bf16*)alloc((size_t)NODES * HID * 2);     // carry, bf16
  __bf16* tb   = (__bf16*)alloc((size_t)NODES * HID * 2);     // block intermediate
  float*  hF   = (float*) alloc((size_t)NODES * HID * 4);     // carry, fp32
  __bf16* uT   = (__bf16*)alloc((size_t)HID * NODES * 2);     // (h@W)^T bf16
  float*  vbuf = (float*) alloc((size_t)NODES * HID * 4);     // h@Wl + b, fp32
  float*  xpad = (float*) alloc((size_t)NODES * 16 * 4);      // padded final output
  float*  part = (float*) alloc((size_t)KSPLIT * NODES * HID * 4); // split-K partials
  __bf16* inWT = (__bf16*)alloc((size_t)HID * INDIM * 2);
  __bf16* inWlT= (__bf16*)alloc((size_t)HID * INDIM * 2);
  __bf16* blkT = (__bf16*)alloc((size_t)LBLK * 4 * HID * HID * 2);
  __bf16* oWT  = (__bf16*)alloc((size_t)16 * HID * 2);
  __bf16* oWlT = (__bf16*)alloc((size_t)16 * HID * 2);
  float*  obp  = (float*) alloc(16 * 4);
  const size_t pStride = (size_t)NODES * HID;

  // ---- precompute: bf16 conversions + transposed weights ----
  {
    long long n4 = (long long)NODES * NODES / 4;
    k_f32_to_bf16_x4<<<dim3((unsigned)((n4 + 255) / 256)), 256, 0, stream>>>(adj, adjb, n4);
    long long m4 = (long long)NODES * INDIM / 4;
    k_f32_to_bf16_x4<<<dim3((unsigned)((m4 + 255) / 256)), 256, 0, stream>>>(x, xb, m4);
  }
  auto tw = [&](const float* s, __bf16* d, int K, int Ns, int Np) {
    int tot = K * Np;
    k_transposeW<<<dim3((tot + 255) / 256), 256, 0, stream>>>(s, d, K, Ns, Np);
  };
  tw(in_W,  inWT,  INDIM, HID, HID);
  tw(in_Wl, inWlT, INDIM, HID, HID);
  const size_t wsz = (size_t)HID * HID;
  for (int l = 0; l < LBLK; ++l) {
    tw(bW1  + l * wsz, blkT + (size_t)(l * 4 + 0) * wsz, HID, HID, HID);
    tw(bWl1 + l * wsz, blkT + (size_t)(l * 4 + 1) * wsz, HID, HID, HID);
    tw(bW2  + l * wsz, blkT + (size_t)(l * 4 + 2) * wsz, HID, HID, HID);
    tw(bWl2 + l * wsz, blkT + (size_t)(l * 4 + 3) * wsz, HID, HID, HID);
  }
  tw(oW,  oWT,  HID, 3, 16);
  tw(oWl, oWlT, HID, 3, 16);
  k_padbias<<<1, 16, 0, stream>>>(ob, obp, 3, 16);

  // ---- wrappers ----
  const dim3 blk(256);
  const dim3 g1d(NODES / 128, 2);                // small GEMMs
  const dim3 g2d(NODES / 256, 2, KSPLIT);        // big GEMM: 256 WGs
  auto g1u = [&](const __bf16* Aact, int K, const __bf16* WT) {   // uT = (A@W)^T bf16
    gemm1_bf16<6><<<g1d, blk, 0, stream>>>(Aact, K, WT, K, nullptr, nullptr, HID, uT, NODES, K);
  };
  auto g1v = [&](const __bf16* Aact, int K, const __bf16* WlT, const float* bias) { // v = A@Wl + b
    gemm1_bf16<6><<<g1d, blk, 0, stream>>>(Aact, K, WlT, K, bias, vbuf, HID, nullptr, 0, K);
  };
  auto g2 = [&](float* oF, __bf16* oB, const float* hres, int relu) { // adj@u + v, epilogue
    gemm2_part<6><<<g2d, blk, 0, stream>>>(adjb, NODES, uT, NODES, part, pStride, HID, NODES);
    int total = NODES * HID;
    k_combine<<<dim3((total + 255) / 256), 256, 0, stream>>>(
        part, pStride, vbuf, hres, oF, oB, total, relu);
  };

  // ---- network ----
  // h0 = relu(adj@(x@inW) + x@inWl + b)
  g1u(xb, INDIM, inWT);
  g1v(xb, INDIM, inWlT, in_b);
  g2(hF, hb, nullptr, 1);

  for (int l = 0; l < LBLK; ++l) {
    const __bf16* W1T  = blkT + (size_t)(l * 4 + 0) * wsz;
    const __bf16* Wl1T = blkT + (size_t)(l * 4 + 1) * wsz;
    const __bf16* W2T  = blkT + (size_t)(l * 4 + 2) * wsz;
    const __bf16* Wl2T = blkT + (size_t)(l * 4 + 3) * wsz;
    // t = relu(gconv(h))
    g1u(hb, HID, W1T);
    g1v(hb, HID, Wl1T, bb1 + (size_t)l * HID);
    g2(nullptr, tb, nullptr, 1);
    // h = (h + relu(gconv(t))) * 0.5
    g1u(tb, HID, W2T);
    g1v(tb, HID, Wl2T, bb2 + (size_t)l * HID);
    g2(hF, hb, hF, 1);
  }

  // x_out = gconv(h) with N padded 3 -> 16 (no relu)
  gemm1_bf16<1><<<dim3(NODES / 128, 1), blk, 0, stream>>>(
      hb, HID, oWT, HID, nullptr, nullptr, 16, uT, NODES, HID);
  gemm1_bf16<1><<<dim3(NODES / 128, 1), blk, 0, stream>>>(
      hb, HID, oWlT, HID, obp, vbuf, 16, nullptr, 0, HID);
  gemm2_part<1><<<dim3(NODES / 256, 1, KSPLIT), blk, 0, stream>>>(
      adjb, NODES, uT, NODES, part, pStride, 16, NODES);
  {
    int total = NODES * 16;
    k_combine<<<dim3((total + 255) / 256), 256, 0, stream>>>(
        part, pStride, vbuf, nullptr, xpad, nullptr, total, 0);
  }

  // pack outputs: [x_out (8192x3) | h (8192x192)]
  int tot = NODES * 3 + NODES * HID;
  k_pack_out<<<dim3((tot + 255) / 256), 256, 0, stream>>>(xpad, hF, out);
}
